// GlobalCluster_14740327760104
// MI455X (gfx1250) — matrix operand order repaired and checked by reading, not compiled, not run on hardware
//
#include <hip/hip_runtime.h>
#include <math.h>

// ---- problem constants (match reference) ----
#define N_B   4
#define L_Q   2048
#define S_K   2048
#define D_IN  256
#define HDIM  256
#define FCH   8
#define CH    32              // per-head channel count (HDIM/FCH)
#define MH    (N_B * FCH)     // 32 flattened heads
#define EPSN  1e-12f

typedef __attribute__((ext_vector_type(16))) _Float16 v16h;
typedef __attribute__((ext_vector_type(8)))  float    v8f;
typedef __attribute__((ext_vector_type(4)))  unsigned int u32x4;
typedef __attribute__((ext_vector_type(8)))  int      i32x8;
typedef __attribute__((ext_vector_type(4)))  int      i32x4;

#if __has_builtin(__builtin_amdgcn_tensor_load_to_lds) && __has_builtin(__builtin_amdgcn_s_wait_tensorcnt)
#define USE_TDM 1
#endif

// ------------------------------------------------------------------
// Kernel 1: x0p = x0 @ W0 + b0, per-head L2 normalize, pack f16
// ------------------------------------------------------------------
__global__ __launch_bounds__(256) void proj0_kernel(
    const float* __restrict__ x0, const float* __restrict__ W0,
    const float* __restrict__ b0, _Float16* __restrict__ x0n)
{
    __shared__ float xs[D_IN];
    const int row = blockIdx.x;              // n*L + l
    const int n = row / L_Q, l = row % L_Q;
    const int j = threadIdx.x;
    xs[j] = x0[(size_t)row * D_IN + j];
    __syncthreads();
    float acc = b0[j];
    #pragma unroll 8
    for (int k = 0; k < D_IN; ++k)
        acc = fmaf(xs[k], W0[(size_t)k * HDIM + j], acc);
    float ss = acc * acc;                    // wave32 == one head's 32 channels
    #pragma unroll
    for (int off = 16; off >= 1; off >>= 1)
        ss += __shfl_xor(ss, off, 32);
    const float inv = 1.0f / fmaxf(sqrtf(ss), EPSN);
    const int head = j >> 5, c = j & 31;
    const int m = n * FCH + head;
    x0n[((size_t)m * L_Q + l) * CH + c] = (_Float16)(acc * inv);
}

// ------------------------------------------------------------------
// Kernel 2: x1p = x1 @ W1 + b1 ; split point/value per head.
// ------------------------------------------------------------------
__global__ __launch_bounds__(512) void proj1_kernel(
    const float* __restrict__ x1, const float* __restrict__ W1,
    const float* __restrict__ b1, _Float16* __restrict__ x1n,
    float* __restrict__ x1v)
{
    __shared__ float xs[D_IN];
    const int row = blockIdx.x;              // n*S + s
    const int n = row / S_K, s = row % S_K;
    const int j = threadIdx.x;
    if (j < D_IN) xs[j] = x1[(size_t)row * D_IN + j];
    __syncthreads();
    float acc = b1[j];
    #pragma unroll 8
    for (int k = 0; k < D_IN; ++k)
        acc = fmaf(xs[k], W1[(size_t)k * (2 * HDIM) + j], acc);
    const int head = j >> 6, ch = j & 63;
    const int m = n * FCH + head;
    if (ch < CH) {                           // point half: wave-uniform branch
        float ss = acc * acc;
        #pragma unroll
        for (int off = 16; off >= 1; off >>= 1)
            ss += __shfl_xor(ss, off, 32);
        const float inv = 1.0f / fmaxf(sqrtf(ss), EPSN);
        x1n[((size_t)m * S_K + s) * CH + ch] = (_Float16)(acc * inv);
    } else {
        x1v[((size_t)m * S_K + s) * CH + (ch - CH)] = acc;
    }
}

// ------------------------------------------------------------------
// TDM helper: DMA one 16-row x 64B B-tile (2D tensor, 4B units) to LDS
// D# layout per CDNA5 ISA 8.3/8.4.  6-arg builtin (clang-23 variant).
// ------------------------------------------------------------------
#ifdef USE_TDM
__device__ __forceinline__ void tdm_load_tile(const _Float16* gsrc, unsigned int lds_off)
{
    const unsigned long long ga = (unsigned long long)(uintptr_t)gsrc;
    u32x4 g0;
    g0[0] = 1u;                                   // count=1, user mode
    g0[1] = lds_off;                              // lds_addr (bytes)
    g0[2] = (unsigned int)(ga & 0xFFFFFFFFull);   // global_addr[31:0]
    g0[3] = (unsigned int)((ga >> 32) & 0x1FFFFFFull) | (2u << 30);  // addr[56:32] | type=2
    i32x8 g1;
    g1[0] = (2 << 16);          // workgroup_mask=0, data_size=2 (4B units)
    g1[1] = (16 << 16);         // tensor_dim0 = 16 dwords per row
    g1[2] = (S_K << 16);        // tensor_dim1 = 2048 rows
    g1[3] = (16 << 16);         // tile_dim0 = 16 dwords
    g1[4] = 16;                 // tile_dim1 = 16 rows, tile_dim2 = 0
    g1[5] = 16;                 // tensor_dim0_stride = 16 dwords
    g1[6] = 0;
    g1[7] = 0;
    const i32x4 z4 = {0, 0, 0, 0};
    const i32x8 z8 = {0, 0, 0, 0, 0, 0, 0, 0};
    __builtin_amdgcn_tensor_load_to_lds(g0, g1, z4, z4, z8, 0);
}
#endif

// ------------------------------------------------------------------
// Kernel 3: fused sim GEMM (V_WMMA_F32_16X16X32_F16) + max/argmax.
// Block = 4 waves, one head, 8 consecutive l-tiles (2 per wave).
// B tiles double-buffered in LDS via the Tensor Data Mover; all 4
// waves consume the same DMA'd tile. sim never touches memory.
// alpha sign folded into the A fragment so the scan tracks raw WMMA
// output (monotone sigmoid∘affine applied once per row at the end).
// ------------------------------------------------------------------
__global__ __launch_bounds__(128) void sim_max_kernel(
    const _Float16* __restrict__ x0n, const _Float16* __restrict__ x1n,
    const float* __restrict__ alpha, const float* __restrict__ beta,
    float* __restrict__ maxval, int* __restrict__ maxidx)
{
    __shared__ __align__(64) _Float16 bs[2][16 * CH];   // 2 x 1KB B tiles

    const int wave = threadIdx.x >> 5;
    const int lane = threadIdx.x & 31;
    const int m     = blockIdx.x >> 4;                  // 16 blocks per head
    const int l0    = (blockIdx.x & 15) * 128 + wave * 32;  // 2 tiles per wave

    const int nrow  = lane & 15;                        // A row / B col / D col
    const int kbase = (lane < 16) ? 0 : 16;             // A K-base (halves, contiguous)
    const int bofs  = (lane < 16) ? 0 : 16;             // B K-base (halves)

    const float al = alpha[0], be = beta[0];
    const float aal = fabsf(al);

    // Two resident A fragments (rows l0..l0+15 and l0+16..l0+31)
    const _Float16* arow0 = x0n + ((size_t)m * L_Q + l0 + nrow) * CH;
    const _Float16* arow1 = arow0 + (size_t)16 * CH;
    v16h a0, a1;
    #pragma unroll
    for (int t = 0; t < 8; ++t) {
        a0[t] = arow0[kbase + t];  a0[8 + t] = arow0[kbase + 8 + t];
        a1[t] = arow1[kbase + t];  a1[8 + t] = arow1[kbase + 8 + t];
    }
    if (al < 0.0f) {                 // fold alpha sign into A (wave-uniform)
        #pragma unroll
        for (int t = 0; t < 16; ++t) { a0[t] = -a0[t]; a1[t] = -a1[t]; }
    }

    float rmax0[8], rmax1[8];
    int   ridx0[8], ridx1[8];
    #pragma unroll
    for (int r = 0; r < 8; ++r) {
        rmax0[r] = -__builtin_inff(); ridx0[r] = 0;
        rmax1[r] = -__builtin_inff(); ridx1[r] = 0;
    }

    const _Float16* bbase = x1n + (size_t)m * S_K * CH;
#ifdef USE_TDM
    const unsigned int lds_base = (unsigned int)(uintptr_t)(void*)&bs[0][0];
    if (wave == 0) tdm_load_tile(bbase, lds_base);          // prologue: buf0
    __builtin_amdgcn_s_wait_tensorcnt(0);                   // no-op for waves 1-3
    __syncthreads();
#endif

    int cur = 0;
    for (int s0 = 0; s0 < S_K; s0 += 16) {
#ifdef USE_TDM
        const int nxt = cur ^ 1;
        if (wave == 0 && s0 + 16 < S_K)                      // prefetch next tile
            tdm_load_tile(bbase + (size_t)(s0 + 16) * CH, lds_base + (unsigned)(nxt * 16 * CH * 2));
        v16h b = *(const v16h*)&bs[cur][(size_t)nrow * CH + bofs];
#else
        v16h b = *(const v16h*)(bbase + (size_t)(s0 + nrow) * CH + bofs);
#endif
        v8f c0 = {0.f, 0.f, 0.f, 0.f, 0.f, 0.f, 0.f, 0.f};
        c0 = __builtin_amdgcn_wmma_f32_16x16x32_f16(false, a0, false, b, (short)0, c0, false, false);
        v8f c1 = {0.f, 0.f, 0.f, 0.f, 0.f, 0.f, 0.f, 0.f};
        c1 = __builtin_amdgcn_wmma_f32_16x16x32_f16(false, a1, false, b, (short)0, c1, false, false);

        const int sidx = s0 + nrow;              // this lane's s column
        #pragma unroll
        for (int r = 0; r < 8; ++r) {
            if (c0[r] > rmax0[r]) { rmax0[r] = c0[r]; ridx0[r] = sidx; }
            if (c1[r] > rmax1[r]) { rmax1[r] = c1[r]; ridx1[r] = sidx; }
        }
#ifdef USE_TDM
        __builtin_amdgcn_s_wait_tensorcnt(0);
        __syncthreads();
        cur = nxt;
#endif
    }

    // reduce across the 16 lanes sharing each output row (xor stays in-half)
    #pragma unroll
    for (int off = 1; off < 16; off <<= 1) {
        #pragma unroll
        for (int r = 0; r < 8; ++r) {
            float ov = __shfl_xor(rmax0[r], off, 32);
            int   oi = __shfl_xor(ridx0[r], off, 32);
            if (ov > rmax0[r] || (ov == rmax0[r] && oi < ridx0[r])) { rmax0[r] = ov; ridx0[r] = oi; }
            ov = __shfl_xor(rmax1[r], off, 32);
            oi = __shfl_xor(ridx1[r], off, 32);
            if (ov > rmax1[r] || (ov == rmax1[r] && oi < ridx1[r])) { rmax1[r] = ov; ridx1[r] = oi; }
        }
    }

    if (nrow == 0) {                             // lanes 0 / 16 -> rows 0-7 / 8-15
        const int mrow0 = (lane < 16) ? 0 : 8;
        const size_t base = (size_t)m * L_Q + l0;
        #pragma unroll
        for (int r = 0; r < 8; ++r) {
            float t0 = fmaf(aal, rmax0[r], be);  // |alpha|*max(sgn*sim)+beta
            float t1 = fmaf(aal, rmax1[r], be);
            maxval[base + mrow0 + r]      = 1.0f / (1.0f + expf(-t0));
            maxidx[base + mrow0 + r]      = ridx0[r];
            maxval[base + 16 + mrow0 + r] = 1.0f / (1.0f + expf(-t1));
            maxidx[base + 16 + mrow0 + r] = ridx1[r];
        }
    }
}

// ------------------------------------------------------------------
// Kernel 4: gather x1_value at argmax, scale, out = new_x0 @ Wm + bm
// ------------------------------------------------------------------
__global__ __launch_bounds__(256) void out_kernel(
    const float* __restrict__ x1v, const float* __restrict__ maxval,
    const int* __restrict__ maxidx, const float* __restrict__ Wm,
    const float* __restrict__ bm, float* __restrict__ out)
{
    __shared__ float nx[HDIM];
    const int row = blockIdx.x;              // n*L + l
    const int n = row / L_Q, l = row % L_Q;
    const int j = threadIdx.x;
    const int head = j >> 5, c = j & 31;
    const int m = n * FCH + head;
    const size_t ml = (size_t)m * L_Q + l;
    const float mv  = maxval[ml];
    const int   idx = maxidx[ml];
    nx[j] = mv * x1v[((size_t)m * S_K + idx) * CH + c];
    __syncthreads();
    float acc = bm[j];
    #pragma unroll 8
    for (int k = 0; k < HDIM; ++k)
        acc = fmaf(nx[k], Wm[(size_t)k * D_IN + j], acc);
    out[(size_t)row * D_IN + j] = acc;
}

// ------------------------------------------------------------------
extern "C" void kernel_launch(void* const* d_in, const int* in_sizes, int n_in,
                              void* d_out, int out_size, void* d_ws, size_t ws_size,
                              hipStream_t stream)
{
    const float* x0    = (const float*)d_in[0];
    const float* x1    = (const float*)d_in[1];
    const float* W0    = (const float*)d_in[2];
    const float* b0    = (const float*)d_in[3];
    const float* W1    = (const float*)d_in[4];
    const float* b1    = (const float*)d_in[5];
    const float* Wm    = (const float*)d_in[6];
    const float* bm    = (const float*)d_in[7];
    const float* alpha = (const float*)d_in[8];
    const float* beta  = (const float*)d_in[9];
    float* out = (float*)d_out;

    char* ws = (char*)d_ws;
    _Float16* x0n = (_Float16*)(ws);                       // 4 MiB
    _Float16* x1n = (_Float16*)(ws + ((size_t)4  << 20));  // 4 MiB
    float*    x1v = (float*)   (ws + ((size_t)8  << 20));  // 8 MiB
    float*    mv  = (float*)   (ws + ((size_t)16 << 20));  // 256 KiB
    int*      mi  = (int*)     (ws + ((size_t)16 << 20) + ((size_t)256 << 10));

    proj0_kernel<<<N_B * L_Q, 256, 0, stream>>>(x0, W0, b0, x0n);
    proj1_kernel<<<N_B * S_K, 512, 0, stream>>>(x1, W1, b1, x1n, x1v);
    sim_max_kernel<<<MH * 16, 128, 0, stream>>>(x0n, x1n, alpha, beta, mv, mi);
    out_kernel<<<N_B * L_Q, 256, 0, stream>>>(x1v, mv, mi, Wm, bm, out);
}
